// Arcfacelayer_9672266350736
// MI455X (gfx1250) — matrix operand order repaired and checked by reading, not compiled, not run on hardware
//
#include <hip/hip_runtime.h>
#include <hip/hip_bf16.h>

// ---------------- CDNA5 WMMA types ----------------
typedef __attribute__((ext_vector_type(16))) __bf16 v16bf;
typedef __attribute__((ext_vector_type(8)))  __bf16 v8bf;
typedef __attribute__((ext_vector_type(8)))  float  v8f;

#define CONCAT16(a, b) __builtin_shufflevector((a), (b), \
    0,1,2,3,4,5,6,7,8,9,10,11,12,13,14,15)

// ---------------- Problem constants ----------------
#define M_DIM   2048
#define K_DIM   512
#define N_DIM   10572
#define N_PAD   10624          // 166 super-tiles * 64 columns
#define NT      4              // N tiles (16 cols each) per wave
#define MT      4              // M tiles (16 rows each) per wave
#define NSUPER  166            // N_PAD / 64
#define MSUPER  32             // M_DIM / 64 (4 M-tiles per wave)
#define RCHUNKS 8              // row chunks for B column-norm reduction

// ArcFace constants (S=64, M=0.5)
#define ARC_S      64.0f
#define ARC_COS_M  0.8775825618903728f
#define ARC_SIN_M  0.479425538604203f
#define ARC_MM     0.2397127693021015f   // sin(M)*M
#define ARC_TH    (-0.8775825618903728f) // cos(pi - M)

// ---------------- bf16 split helpers ----------------
__device__ __forceinline__ unsigned short f32_to_bf16_rne(float f) {
    unsigned int u = __float_as_uint(f);
    unsigned int r = 0x7FFFu + ((u >> 16) & 1u);
    return (unsigned short)((u + r) >> 16);
}
__device__ __forceinline__ float bf16us_to_f32(unsigned short h) {
    return __uint_as_float(((unsigned int)h) << 16);
}

// =====================================================================
// Kernel 1: L2-normalize embedding rows, split into bf16 hi/lo.
// One wave per row (wave32); each lane handles 16 contiguous elements.
// =====================================================================
__global__ void prep_a_kernel(const float* __restrict__ emb,
                              unsigned short* __restrict__ ahi,
                              unsigned short* __restrict__ alo) {
    int wave = blockIdx.x * (blockDim.x >> 5) + (threadIdx.x >> 5);
    int lane = threadIdx.x & 31;
    if (wave >= M_DIM) return;

    const float* row = emb + (size_t)wave * K_DIM + lane * 16;
    float v[16];
    float s = 0.f;
#pragma unroll
    for (int i = 0; i < 16; ++i) { v[i] = row[i]; s += v[i] * v[i]; }
#pragma unroll
    for (int m = 16; m >= 1; m >>= 1) s += __shfl_xor(s, m, 32);
    float inv = 1.0f / sqrtf(s);

    size_t o = (size_t)wave * K_DIM + lane * 16;
#pragma unroll
    for (int i = 0; i < 16; ++i) {
        float x = v[i] * inv;
        unsigned short h = f32_to_bf16_rne(x);
        ahi[o + i] = h;
        alo[o + i] = f32_to_bf16_rne(x - bf16us_to_f32(h));
    }
}

// =====================================================================
// Kernel 2a: partial sum-of-squares per column over a 64-row chunk.
// Grid (col blocks, RCHUNKS); deterministic (no atomics).
// =====================================================================
__global__ void prep_b_norm1_kernel(const float* __restrict__ ker,
                                    float* __restrict__ partial) {
    int j = blockIdx.x * blockDim.x + threadIdx.x;
    int chunk = blockIdx.y;
    if (j >= N_PAD) return;
    float s = 0.f;
    if (j < N_DIM) {
        int r0 = chunk * (K_DIM / RCHUNKS);
#pragma unroll 4
        for (int r = 0; r < K_DIM / RCHUNKS; ++r) {
            float v = ker[(size_t)(r0 + r) * N_DIM + j];
            s += v * v;
        }
    }
    partial[(size_t)chunk * N_PAD + j] = s;
}

// Kernel 2b: finalize inverse norms.
__global__ void prep_b_norm2_kernel(const float* __restrict__ partial,
                                    float* __restrict__ cinv) {
    int j = blockIdx.x * blockDim.x + threadIdx.x;
    if (j >= N_PAD) return;
    float s = 0.f;
#pragma unroll
    for (int c = 0; c < RCHUNKS; ++c) s += partial[(size_t)c * N_PAD + j];
    cinv[j] = (j < N_DIM) ? (1.0f / sqrtf(s)) : 0.f;
}

// =====================================================================
// Kernel 3: transpose + normalize + bf16 hi/lo split:
//   Bt[j][r] = kernel[r][j] * cinv[j],  j in [0, N_PAD), zero padded.
// 16x16 LDS tile transpose so both read and write are coalesced.
// =====================================================================
__global__ void prep_b_split_kernel(const float* __restrict__ ker,
                                    const float* __restrict__ cinv,
                                    unsigned short* __restrict__ bhi,
                                    unsigned short* __restrict__ blo) {
    __shared__ float tile[16][17];
    int j0 = blockIdx.x * 16;      // column tile
    int r0 = blockIdx.y * 16;      // row tile
    int tx = threadIdx.x, ty = threadIdx.y;

    int j = j0 + tx;
    float v = 0.f;
    if (j < N_DIM) v = ker[(size_t)(r0 + ty) * N_DIM + j] * cinv[j];
    tile[ty][tx] = v;
    __syncthreads();

    // write Bt[j0+ty][r0+tx] = tile[tx][ty]
    float x = tile[tx][ty];
    size_t o = (size_t)(j0 + ty) * K_DIM + (r0 + tx);
    unsigned short h = f32_to_bf16_rne(x);
    bhi[o] = h;
    blo[o] = f32_to_bf16_rne(x - bf16us_to_f32(h));
}

// =====================================================================
// Kernel 4: split-fp32 (bf16x3) WMMA GEMM + fused ArcFace epilogue.
// One wave per 64x64 output tile (4 M-tiles x 4 N-tiles):
//   - A fragments reused across 4 N tiles
//   - B fragments reused across 4 M tiles
//   48 WMMAs per 32 b128 loads per K-step (48 FLOP per L2 byte).
// acc += Ahi*Bhi + Ahi*Blo + Alo*Bhi (f32 accumulate, C->D chained;
// 16 independent accumulator chains for latency hiding).
// =====================================================================
__global__ void __launch_bounds__(256)
arcface_gemm_kernel(const unsigned short* __restrict__ ahi,
                    const unsigned short* __restrict__ alo,
                    const unsigned short* __restrict__ bhi,
                    const unsigned short* __restrict__ blo,
                    const int*            __restrict__ labels,
                    float*                __restrict__ out) {
    int wave = blockIdx.x * (blockDim.x >> 5) + (threadIdx.x >> 5);
    int lane = threadIdx.x & 31;
    int mt = wave / NSUPER;            // 0..MSUPER-1
    int ns = wave - mt * NSUPER;
    int m0 = mt * (MT * 16);
    int n0 = ns * (NT * 16);

    int lrow  = lane & 15;
    int lhalf = lane >> 4;

    // A fragment addressing (16-bit A 16x32 layout, ISA 7.12.2):
    //   lanes 0-15: K 0..7 in v0-3, K 16..23 in v4-7; lanes 16-31: +8
    size_t abase = (size_t)(m0 + lrow) * K_DIM + (lhalf << 3);
    // B fragment addressing (same K layout as A):
    //   lanes 0-15: K 0..7 then K 16..23; lanes 16-31: +8
    size_t bbase = (size_t)(n0 + lrow) * K_DIM + (lhalf << 3);

    v8f acc[MT][NT] = {};   // f32 accumulators: 4 M tiles x 4 N tiles

    for (int k0 = 0; k0 < K_DIM; k0 += 32) {
        // ---- A fragments (4 M tiles x hi/lo): two b128 loads each ----
        v16bf Ahi[MT], Alo[MT];
#pragma unroll
        for (int mi = 0; mi < MT; ++mi) {
            const v8bf* pah = (const v8bf*)(ahi + abase + (size_t)mi * 16 * K_DIM + k0);
            const v8bf* pal = (const v8bf*)(alo + abase + (size_t)mi * 16 * K_DIM + k0);
            Ahi[mi] = CONCAT16(pah[0], pah[2]);
            Alo[mi] = CONCAT16(pal[0], pal[2]);
        }

#pragma unroll
        for (int t = 0; t < NT; ++t) {
            const v8bf* pbh = (const v8bf*)(bhi + bbase + (size_t)t * 16 * K_DIM + k0);
            const v8bf* pbl = (const v8bf*)(blo + bbase + (size_t)t * 16 * K_DIM + k0);
            v16bf Bhi = CONCAT16(pbh[0], pbh[2]);
            v16bf Blo = CONCAT16(pbl[0], pbl[2]);

#pragma unroll
            for (int mi = 0; mi < MT; ++mi) {
                acc[mi][t] = __builtin_amdgcn_wmma_f32_16x16x32_bf16(
                    false, Ahi[mi], false, Bhi, (short)0, acc[mi][t], false, false);
                acc[mi][t] = __builtin_amdgcn_wmma_f32_16x16x32_bf16(
                    false, Ahi[mi], false, Blo, (short)0, acc[mi][t], false, false);
                acc[mi][t] = __builtin_amdgcn_wmma_f32_16x16x32_bf16(
                    false, Alo[mi], false, Bhi, (short)0, acc[mi][t], false, false);
            }
        }
    }

    // ---- ArcFace epilogue ----
    // C/D layout: VGPR i -> row mbase + i + 8*lhalf; column = n tile base + (lane&15)
    int labv[MT][8];
#pragma unroll
    for (int mi = 0; mi < MT; ++mi)
#pragma unroll
        for (int i = 0; i < 8; ++i)
            labv[mi][i] = labels[m0 + mi * 16 + (lhalf << 3) + i];

#pragma unroll
    for (int t = 0; t < NT; ++t) {
        int n = n0 + t * 16 + lrow;
        if (n < N_DIM) {
#pragma unroll
            for (int mi = 0; mi < MT; ++mi) {
#pragma unroll
                for (int i = 0; i < 8; ++i) {
                    float c = acc[mi][t][i];         // cos_t (pre-normalized inputs)
                    float s = sqrtf(1.0f - c * c);
                    float cos_mt = ARC_S * (c * ARC_COS_M - s * ARC_SIN_M);
                    float keep   = ARC_S * (c - ARC_MM);
                    float sel    = (c > ARC_TH) ? cos_mt : keep;
                    float o      = (labv[mi][i] == n) ? sel : ARC_S * c;
                    int m = m0 + mi * 16 + (lhalf << 3) + i;
                    out[(size_t)m * N_DIM + n] = o;
                }
            }
        }
    }
}

// =====================================================================
// Launch
// =====================================================================
extern "C" void kernel_launch(void* const* d_in, const int* in_sizes, int n_in,
                              void* d_out, int out_size, void* d_ws, size_t ws_size,
                              hipStream_t stream) {
    const float* emb    = (const float*)d_in[0];   // [2048, 512]
    const float* ker    = (const float*)d_in[1];   // [512, 10572]
    const int*   labels = (const int*)d_in[2];     // [2048]
    float*       out    = (float*)d_out;           // [2048, 10572]

    // Workspace layout (~26.3 MB total):
    unsigned short* ahi     = (unsigned short*)d_ws;
    unsigned short* alo     = ahi + (size_t)M_DIM * K_DIM;
    unsigned short* bhi     = alo + (size_t)M_DIM * K_DIM;
    unsigned short* blo     = bhi + (size_t)N_PAD * K_DIM;
    float*          cinv    = (float*)(blo + (size_t)N_PAD * K_DIM);
    float*          partial = cinv + N_PAD;

    // 1) normalize + split A: one wave per row, 8 waves per block
    prep_a_kernel<<<M_DIM / 8, 256, 0, stream>>>(emb, ahi, alo);

    // 2) column inverse norms of B: two-phase parallel reduction
    prep_b_norm1_kernel<<<dim3((N_PAD + 255) / 256, RCHUNKS), 256, 0, stream>>>(
        ker, partial);
    prep_b_norm2_kernel<<<(N_PAD + 255) / 256, 256, 0, stream>>>(partial, cinv);

    // 3) transpose + normalize + split B (zero-padded to N_PAD)
    prep_b_split_kernel<<<dim3(N_PAD / 16, K_DIM / 16), dim3(16, 16), 0, stream>>>(
        ker, cinv, bhi, blo);

    // 4) WMMA GEMM + ArcFace epilogue: 32*166 = 5312 waves, 8 per block
    arcface_gemm_kernel<<<(MSUPER * NSUPER) / 8, 256, 0, stream>>>(
        ahi, alo, bhi, blo, labels, out);
}
